// SAMixer_15659450761973
// MI455X (gfx1250) — matrix operand order repaired and checked by reading, not compiled, not run on hardware
//
#include <hip/hip_runtime.h>

typedef __bf16 bf16;
typedef __attribute__((ext_vector_type(16))) __bf16 v16bf;
typedef __attribute__((ext_vector_type(8)))  __bf16 v8bf;
typedef __attribute__((ext_vector_type(8)))  float  v8f;
typedef __attribute__((ext_vector_type(4)))  int    v4i;

constexpr int   NBA = 2;            // batch
constexpr int   NCH = 128;          // channels C
constexpr int   NGR = 2;            // groups
constexpr int   NDH = 64;           // head dim
constexpr int   H0 = 28,  W0 = 28;  // x1 grid
constexpr int   H1 = 112, W1v = 112;
constexpr int   N1 = H1 * W1v;      // 12544 tokens (attention res)
constexpr int   H2 = 224, W2v = 224;
constexpr int   N2 = H2 * W2v;      // 50176 pixels (full res)
constexpr float EPSV = 1e-12f;
constexpr float SCLV = 0.125f;      // D^-0.5

__device__ __forceinline__ float geluf(float x) {
  return 0.5f * x * (1.0f + erff(x * 0.70710678118654752f));
}
__device__ __forceinline__ float wred32(float v) {
  #pragma unroll
  for (int o = 16; o; o >>= 1) v += __shfl_xor(v, o, 32);
  return v;
}

// ---------------- bilinear up x2 + residual add -----------------------------
__global__ void k_upbi_add(const float* __restrict__ in, const float* __restrict__ add,
                           float* __restrict__ out, int BC, int hi, int wi) {
  int ho = hi * 2, wo = wi * 2;
  long total = (long)BC * ho * wo;
  long i = (long)blockIdx.x * blockDim.x + threadIdx.x;
  if (i >= total) return;
  int xo = (int)(i % wo); long t = i / wo;
  int yo = (int)(t % ho); int bc = (int)(t / ho);
  float fy = 0.5f * yo - 0.25f, fx = 0.5f * xo - 0.25f;
  int y0 = (int)floorf(fy), x0 = (int)floorf(fx);
  float wy = fy - y0, wx = fx - x0;
  int y0c = min(max(y0, 0), hi - 1), y1c = min(max(y0 + 1, 0), hi - 1);
  int x0c = min(max(x0, 0), wi - 1), x1c = min(max(x0 + 1, 0), wi - 1);
  const float* p = in + (long)bc * hi * wi;
  float v00 = p[(long)y0c * wi + x0c], v01 = p[(long)y0c * wi + x1c];
  float v10 = p[(long)y1c * wi + x0c], v11 = p[(long)y1c * wi + x1c];
  float v = (1.f - wy) * ((1.f - wx) * v00 + wx * v01) +
            wy * ((1.f - wx) * v10 + wx * v11);
  out[i] = v + add[i];
}

// ---------------- depthwise 2x2 stride2 VALID -------------------------------
__global__ void k_dwconv2(const float* __restrict__ in, const float* __restrict__ w,
                          float* __restrict__ out, int hi, int wi) {
  int ho = hi >> 1, wo = wi >> 1;
  long total = (long)NBA * NCH * ho * wo;
  long i = (long)blockIdx.x * blockDim.x + threadIdx.x;
  if (i >= total) return;
  int xo = (int)(i % wo); long t = i / wo;
  int yo = (int)(t % ho); int bc = (int)(t / ho);
  int c = bc % NCH;
  const float* p = in + (long)bc * hi * wi + (long)(2 * yo) * wi + 2 * xo;
  const float* k = w + c * 4;
  out[i] = p[0] * k[0] + p[1] * k[1] + p[wi] * k[2] + p[wi + 1] * k[3];
}

// ---------------- depthwise 3x3 stride1 pad1 --------------------------------
__global__ void k_dwconv3(const float* __restrict__ in, const float* __restrict__ w,
                          float* __restrict__ out, int hi, int wi) {
  long total = (long)NBA * NCH * hi * wi;
  long i = (long)blockIdx.x * blockDim.x + threadIdx.x;
  if (i >= total) return;
  int xo = (int)(i % wi); long t = i / wi;
  int yo = (int)(t % hi); int bc = (int)(t / hi);
  int c = bc % NCH;
  const float* p = in + (long)bc * hi * wi;
  const float* k = w + c * 9;
  float s = 0.f;
  #pragma unroll
  for (int ky = 0; ky < 3; ky++) {
    int y = yo + ky - 1;
    if (y < 0 || y >= hi) continue;
    #pragma unroll
    for (int kx = 0; kx < 3; kx++) {
      int x = xo + kx - 1;
      if (x < 0 || x >= wi) continue;
      s += p[(long)y * wi + x] * k[ky * 3 + kx];
    }
  }
  out[i] = s;
}

// ------------- per-(b,c) plane stats (NCHW) ---------------------------------
// mode 0: mean, rsqrt(popvar+eps)   mode 1: mean, 1/(sample_std+eps)
__global__ void k_planestat(const float* __restrict__ x, float* __restrict__ stat,
                            int plane, int mode) {
  __shared__ float s1[256], s2[256];
  long base = (long)blockIdx.x * plane;
  float a = 0.f, b = 0.f;
  for (int i = threadIdx.x; i < plane; i += 256) {
    float v = x[base + i]; a += v; b += v * v;
  }
  s1[threadIdx.x] = a; s2[threadIdx.x] = b;
  __syncthreads();
  for (int s = 128; s > 0; s >>= 1) {
    if ((int)threadIdx.x < s) { s1[threadIdx.x] += s1[threadIdx.x + s]; s2[threadIdx.x] += s2[threadIdx.x + s]; }
    __syncthreads();
  }
  if (threadIdx.x == 0) {
    float m = s1[0] / plane;
    float ss = fmaxf(s2[0] - plane * m * m, 0.f);
    float sc = (mode == 0) ? rsqrtf(ss / plane + EPSV)
                           : 1.0f / (sqrtf(ss / (plane - 1)) + EPSV);
    stat[blockIdx.x * 2] = m; stat[blockIdx.x * 2 + 1] = sc;
  }
}

// ------------- per-column stats (token-major) -------------------------------
__global__ void k_colstat(const float* __restrict__ tok, float* __restrict__ stat,
                          int rows, int Cc, int mode) {
  __shared__ float s1[256], s2[256];
  int c = blockIdx.x % Cc; int grp = blockIdx.x / Cc;
  const float* p = tok + (long)grp * rows * Cc + c;
  float a = 0.f, b = 0.f;
  for (int i = threadIdx.x; i < rows; i += 256) {
    float v = p[(long)i * Cc]; a += v; b += v * v;
  }
  s1[threadIdx.x] = a; s2[threadIdx.x] = b;
  __syncthreads();
  for (int s = 128; s > 0; s >>= 1) {
    if ((int)threadIdx.x < s) { s1[threadIdx.x] += s1[threadIdx.x + s]; s2[threadIdx.x] += s2[threadIdx.x + s]; }
    __syncthreads();
  }
  if (threadIdx.x == 0) {
    float m = s1[0] / rows;
    float ss = fmaxf(s2[0] - rows * m * m, 0.f);
    float sc = (mode == 0) ? rsqrtf(ss / rows + EPSV)
                           : 1.0f / (sqrtf(ss / (rows - 1)) + EPSV);
    stat[blockIdx.x * 2] = m; stat[blockIdx.x * 2 + 1] = sc;
  }
}

// ------------- NCHW -> token-major bf16 with groupnorm applied --------------
__global__ void k_gnapply_tok(const float* __restrict__ x, const float* __restrict__ stat,
                              const float* __restrict__ g, const float* __restrict__ bta,
                              bf16* __restrict__ outB, int plane) {
  long total = (long)NBA * NCH * plane;
  long i = (long)blockIdx.x * blockDim.x + threadIdx.x;
  if (i >= total) return;
  int p = (int)(i % plane); int bc = (int)(i / plane);
  int c = bc % NCH; int b = bc / NCH;
  float m = stat[bc * 2], sc = stat[bc * 2 + 1];
  float v = (x[i] - m) * sc * g[c] + bta[c];
  outB[((long)b * plane + p) * NCH + c] = (bf16)v;
}

// ------------- NCHW -> token-major bf16 (plain) -----------------------------
__global__ void k_tokbf(const float* __restrict__ x, bf16* __restrict__ outB, int plane) {
  long total = (long)NBA * NCH * plane;
  long i = (long)blockIdx.x * blockDim.x + threadIdx.x;
  if (i >= total) return;
  int p = (int)(i % plane); int bc = (int)(i / plane);
  int c = bc % NCH; int b = bc / NCH;
  outB[((long)b * plane + p) * NCH + c] = (bf16)x[i];
}

// ------------- f32 -> bf16 elementwise --------------------------------------
__global__ void k_cvt_bf16(const float* __restrict__ in, bf16* __restrict__ out, long n) {
  long i = (long)blockIdx.x * blockDim.x + threadIdx.x;
  if (i < n) out[i] = (bf16)in[i];
}

// ------------- row layernorm on token-major f32 -----------------------------
// mode 0: (x-m)*rsqrt(popvar+eps)*g+b     mode 1: (x-m)/(sample_std+eps)*g+b
__global__ void k_rowln(const float* __restrict__ tok, const float* __restrict__ g,
                        const float* __restrict__ bta, float* __restrict__ outF,
                        bf16* __restrict__ outB, int M, int Cc, int mode) {
  int wave = threadIdx.x >> 5, lane = threadIdx.x & 31;
  long row = (long)blockIdx.x * 8 + wave;
  if (row >= M) return;
  const float* p = tok + row * Cc;
  int per = Cc >> 5;
  float v[8], s = 0.f;
  for (int j = 0; j < per; j++) { v[j] = p[lane + (j << 5)]; s += v[j]; }
  s = wred32(s);
  float m = s / Cc;
  float q = 0.f;
  for (int j = 0; j < per; j++) { float d = v[j] - m; q += d * d; }
  q = wred32(q);
  float sc = (mode == 0) ? rsqrtf(q / Cc + EPSV)
                         : 1.0f / (sqrtf(q / (Cc - 1)) + EPSV);
  for (int j = 0; j < per; j++) {
    int c = lane + (j << 5);
    float o = (v[j] - m) * sc * g[c] + bta[c];
    if (outF) outF[row * Cc + c] = o;
    if (outB) outB[row * Cc + c] = (bf16)o;
  }
}

// ------------- per-pixel channel norm on NCHW -------------------------------
// outputs: optional f32 NCHW, optional token-major bf16
__global__ void k_pixln(const float* __restrict__ x, const float* __restrict__ g,
                        const float* __restrict__ bta, float* __restrict__ outF,
                        bf16* __restrict__ outB, int plane, int mode) {
  int wave = threadIdx.x >> 5, lane = threadIdx.x & 31;
  long pix = (long)blockIdx.x * 8 + wave;
  long total = (long)NBA * plane;
  if (pix >= total) return;
  int b = (int)(pix / plane); int p = (int)(pix % plane);
  const float* basep = x + ((long)b * NCH) * plane + p;
  float v[4], s = 0.f;
  #pragma unroll
  for (int j = 0; j < 4; j++) { v[j] = basep[(long)(lane + j * 32) * plane]; s += v[j]; }
  s = wred32(s);
  float m = s / NCH;
  float q = 0.f;
  #pragma unroll
  for (int j = 0; j < 4; j++) { float d = v[j] - m; q += d * d; }
  q = wred32(q);
  float sc = (mode == 0) ? rsqrtf(q / NCH + EPSV)
                         : 1.0f / (sqrtf(q / (NCH - 1)) + EPSV);
  #pragma unroll
  for (int j = 0; j < 4; j++) {
    int c = lane + j * 32;
    float o = (v[j] - m) * sc * g[c] + bta[c];
    if (outF) outF[((long)b * NCH + c) * plane + p] = o;
    if (outB) outB[pix * NCH + c] = (bf16)o;
  }
}

// ------------- WMMA GEMM: out[M][Nn] = A[M][K](bf16) x W[Nn][K](bf16) + bias -
// 256 thr = 8 waves; block tile 128(M) x 64(N); each wave a 32x32 macro tile
// (4 WMMA accumulators). K stepped by 32 with double-buffered LDS staged via
// GLOBAL_LOAD_ASYNC_TO_LDS_B128 (ASYNCcnt), one barrier per K-step.
__global__ void k_gemm(const bf16* __restrict__ A, const bf16* __restrict__ W,
                       const float* __restrict__ bias, float* __restrict__ outF,
                       bf16* __restrict__ outB, int M, int K, int Nn, int doGelu) {
  __shared__ __align__(16) bf16 As[2][128][32];
  __shared__ __align__(16) bf16 Bs[2][64][32];
  int tid = threadIdx.x;
  int wave = tid >> 5, lane = tid & 31;
  int wm = wave >> 1, wn = wave & 1;         // 4 (M) x 2 (N) waves
  long m0 = (long)blockIdx.y * 128;
  int n0 = blockIdx.x * 64;

  v8f zero = {0.f, 0.f, 0.f, 0.f, 0.f, 0.f, 0.f, 0.f};
  v8f acc[2][2] = {{zero, zero}, {zero, zero}};

  // async-stage one K-slice (32 wide) of A (128 rows) and W (64 rows) into LDS
  auto issue = [&](int buf, int kt) {
    int ch = tid * 2;                        // A: 512 16B chunks, 2 per thread
    #pragma unroll
    for (int u = 0; u < 2; u++, ch++) {
      int row = ch >> 2, kk = (ch & 3) << 3;
      unsigned lo = (unsigned)(unsigned long long)(&As[buf][row][kk]);
      const bf16* gp = A + (m0 + row) * (long)K + kt + kk;
      asm volatile("global_load_async_to_lds_b128 %0, %1, off"
                   :: "v"(lo), "v"(gp) : "memory");
    }
    {                                        // B: 256 16B chunks, 1 per thread
      int row = tid >> 2, kk = (tid & 3) << 3;
      unsigned lo = (unsigned)(unsigned long long)(&Bs[buf][row][kk]);
      const bf16* gp = W + (long)(n0 + row) * K + kt + kk;
      asm volatile("global_load_async_to_lds_b128 %0, %1, off"
                   :: "v"(lo), "v"(gp) : "memory");
    }
  };

  issue(0, 0);
  int nk = K >> 5;
  for (int ki = 0; ki < nk; ki++) {
    asm volatile("s_wait_asynccnt 0x0" ::: "memory");
    __syncthreads();
    if (ki + 1 < nk) issue((ki + 1) & 1, (ki + 1) << 5);
    int buf = ki & 1;

    // fragments (ISA striping): A lanes 0-15 K{0..7,16..23}, 16-31 K{8..15,24..31}
    int arow = (wm << 5) + (lane & 15);
    int akb = (lane >> 4) << 3;
    int bcol = (wn << 5) + (lane & 15);
    int bkb = (lane >> 4) << 3;
    v16bf af[2], bfv[2];
    #pragma unroll
    for (int i = 0; i < 2; i++) {
      v8bf a0 = *(const v8bf*)(&As[buf][arow + 16 * i][akb]);
      v8bf a1 = *(const v8bf*)(&As[buf][arow + 16 * i][akb + 16]);
      v8bf b0 = *(const v8bf*)(&Bs[buf][bcol + 16 * i][bkb]);
      v8bf b1 = *(const v8bf*)(&Bs[buf][bcol + 16 * i][bkb + 16]);
      #pragma unroll
      for (int t = 0; t < 8; t++) {
        af[i][t] = a0[t]; af[i][t + 8] = a1[t];
        bfv[i][t] = b0[t]; bfv[i][t + 8] = b1[t];
      }
    }
    #pragma unroll
    for (int i = 0; i < 2; i++)
      #pragma unroll
      for (int j = 0; j < 2; j++)
        acc[i][j] = __builtin_amdgcn_wmma_f32_16x16x32_bf16(
            false, af[i], false, bfv[j], (short)0, acc[i][j], false, false);
  }

  // epilogue: C/D layout -> lane l reg r: m = r + 8*(l>=16), n = l&15
  int nl = lane & 15;
  int mhi = (lane >> 4) << 3;
  #pragma unroll
  for (int i = 0; i < 2; i++) {
    #pragma unroll
    for (int j = 0; j < 2; j++) {
      int n = n0 + (wn << 5) + (j << 4) + nl;
      float bv = bias ? bias[n] : 0.f;
      #pragma unroll
      for (int r = 0; r < 8; r++) {
        long m = m0 + (wm << 5) + (i << 4) + mhi + r;
        if (m < M) {
          float v = acc[i][j][r] + bv;
          if (doGelu) v = geluf(v);
          if (outF) outF[m * Nn + n] = v;
          if (outB) outB[m * Nn + n] = (bf16)v;
        }
      }
    }
  }
}

// ------------- 5-level pyramid attention ------------------------------------
__global__ void k_attn(const float* __restrict__ q, const float* __restrict__ kv0,
                       const float* __restrict__ kv1, const float* __restrict__ kv2,
                       const float* __restrict__ kv3, const float* __restrict__ kv4,
                       float* __restrict__ u) {
  int wave = threadIdx.x >> 5, lane = threadIdx.x & 31;
  long id = (long)blockIdx.x * 8 + wave;  // over B*N1*G
  if (id >= (long)NBA * N1 * NGR) return;
  int g = (int)(id % NGR); long bn = id / NGR;
  int b = (int)(bn / N1); int n = (int)(bn % N1);
  int yy = n / W1v, xx = n % W1v;
  long i1 = (long)b * (H0 * W0) + (yy >> 2) * W0 + (xx >> 2);
  long i2 = (long)b * (4 * H0 * W0) + (yy >> 1) * (2 * W0) + (xx >> 1);
  int d0 = lane * 2;
  int go = g * NDH + d0;
  float q0 = q[bn * NCH + go], q1 = q[bn * NCH + go + 1];
  const float* kvp[5] = { kv0 + bn * (2 * NCH), kv1 + i1 * (2 * NCH),
                          kv2 + i2 * (2 * NCH), kv3 + bn * (2 * NCH),
                          kv4 + bn * (2 * NCH) };
  float s[5];
  #pragma unroll
  for (int p = 0; p < 5; p++) {
    const float* kp = kvp[p] + go;
    float d = q0 * kp[0] + q1 * kp[1];
    s[p] = wred32(d) * SCLV;
  }
  float mx = s[0];
  #pragma unroll
  for (int p = 1; p < 5; p++) mx = fmaxf(mx, s[p]);
  float e[5], se = 0.f;
  #pragma unroll
  for (int p = 0; p < 5; p++) { e[p] = expf(s[p] - mx); se += e[p]; }
  float inv = 1.0f / se;
  float u0 = 0.f, u1 = 0.f;
  #pragma unroll
  for (int p = 0; p < 5; p++) {
    const float* vp = kvp[p] + NCH + go;
    float a = e[p] * inv;
    u0 += a * vp[0]; u1 += a * vp[1];
  }
  u[bn * NCH + go] = u0;
  u[bn * NCH + go + 1] = u1;
}

// ------------- fnorm + 1x1 conv to 4 phase-detail channels ------------------
__global__ void k_pd(const float* __restrict__ up, const float* __restrict__ stat,
                     const float* __restrict__ fn_std, const float* __restrict__ pdw,
                     const float* __restrict__ pdb, float* __restrict__ pdo) {
  int wave = threadIdx.x >> 5, lane = threadIdx.x & 31;
  long tokid = (long)blockIdx.x * 8 + wave;
  if (tokid >= (long)NBA * N1) return;
  int b = (int)(tokid / N1);
  float acc[4] = {0.f, 0.f, 0.f, 0.f};
  #pragma unroll
  for (int j = 0; j < 4; j++) {
    int c = lane + j * 32;
    int si = (b * NCH + c) * 2;
    float fv = (up[tokid * NCH + c] - stat[si]) * stat[si + 1] * fn_std[c];
    #pragma unroll
    for (int t = 0; t < 4; t++) acc[t] += fv * pdw[t * NCH + c];
  }
  #pragma unroll
  for (int t = 0; t < 4; t++) acc[t] = wred32(acc[t]);
  if (lane < 4) pdo[tokid * 4 + lane] = acc[lane] + pdb[lane];
}

// ------------- pixel-shuffle scatter-add: x += (u + pd) ---------------------
__global__ void k_scatter(const float* __restrict__ uptok, const float* __restrict__ pdo,
                          float* __restrict__ x224) {
  long total = (long)NBA * NCH * N1;
  long i = (long)blockIdx.x * blockDim.x + threadIdx.x;
  if (i >= total) return;
  int p = (int)(i % N1); int bc = (int)(i / N1);
  int c = bc % NCH; int b = bc / NCH;
  int yy = p / W1v, xx = p % W1v;
  float uv = uptok[((long)b * N1 + p) * NCH + c];
  const float* pp = pdo + ((long)b * N1 + p) * 4;
  float* xb = x224 + (long)bc * N2;
  long o = (long)(2 * yy) * W2v + 2 * xx;
  xb[o] += uv + pp[0];
  xb[o + 1] += uv + pp[1];
  xb[o + W2v] += uv + pp[2];
  xb[o + W2v + 1] += uv + pp[3];
}

// ------------- NCHW += token-major f32 --------------------------------------
__global__ void k_addtok(float* __restrict__ x, const float* __restrict__ tok, int plane) {
  long total = (long)NBA * NCH * plane;
  long i = (long)blockIdx.x * blockDim.x + threadIdx.x;
  if (i >= total) return;
  int p = (int)(i % plane); int bc = (int)(i / plane);
  int c = bc % NCH; int b = bc / NCH;
  x[i] += tok[((long)b * plane + p) * NCH + c];
}

// ------------- final batchnorm + gelu (token-major -> NCHW out) -------------
__global__ void k_bnfinal(const float* __restrict__ z, const float* __restrict__ stat,
                          const float* __restrict__ g, const float* __restrict__ bta,
                          float* __restrict__ out, int plane) {
  long total = (long)NBA * NCH * plane;
  long i = (long)blockIdx.x * blockDim.x + threadIdx.x;
  if (i >= total) return;
  int p = (int)(i % plane); int bc = (int)(i / plane);
  int c = bc % NCH; int b = bc / NCH;
  float v = z[((long)b * plane + p) * NCH + c];
  v = (v - stat[c * 2]) * stat[c * 2 + 1] * g[c] + bta[c];
  out[i] = geluf(v);
}

// ============================================================================
extern "C" void kernel_launch(void* const* d_in, const int* in_sizes, int n_in,
                              void* d_out, int out_size, void* d_ws, size_t ws_size,
                              hipStream_t stream) {
  (void)in_sizes; (void)n_in; (void)out_size; (void)ws_size;
  const float* x1 = (const float*)d_in[0];
  const float* x2 = (const float*)d_in[1];
  const float* x3 = (const float*)d_in[2];
  const float* x4 = (const float*)d_in[3];
  const float* al1_dw = (const float*)d_in[4];
  const float* al1_gng = (const float*)d_in[5];
  const float* al1_gnb = (const float*)d_in[6];
  const float* al1_pw = (const float*)d_in[7];
  const float* al1_pwb = (const float*)d_in[8];
  const float* al2_dw = (const float*)d_in[9];
  const float* al2_gng = (const float*)d_in[10];
  const float* al2_gnb = (const float*)d_in[11];
  const float* al2_pw = (const float*)d_in[12];
  const float* al2_pwb = (const float*)d_in[13];
  const float* ln0_g = (const float*)d_in[14]; const float* ln0_b = (const float*)d_in[15];
  const float* ln1_g = (const float*)d_in[16]; const float* ln1_b = (const float*)d_in[17];
  const float* ln2_g = (const float*)d_in[18]; const float* ln2_b = (const float*)d_in[19];
  const float* ln3_g = (const float*)d_in[20]; const float* ln3_b = (const float*)d_in[21];
  const float* ln4_g = (const float*)d_in[22]; const float* ln4_b = (const float*)d_in[23];
  const float* wq = (const float*)d_in[24];  const float* bq = (const float*)d_in[25];
  const float* wkv = (const float*)d_in[26]; const float* bkv = (const float*)d_in[27];
  const float* wp = (const float*)d_in[28];  const float* bp = (const float*)d_in[29];
  const float* fn_std = (const float*)d_in[30];
  const float* pd_w = (const float*)d_in[31]; const float* pd_b = (const float*)d_in[32];
  const float* gi_g = (const float*)d_in[33]; const float* gi_b = (const float*)d_in[34];
  const float* mlp_dw = (const float*)d_in[35];
  const float* mlp_gng = (const float*)d_in[36]; const float* mlp_gnb = (const float*)d_in[37];
  const float* mlp_w1 = (const float*)d_in[38]; const float* mlp_b1 = (const float*)d_in[39];
  const float* mlp_w2 = (const float*)d_in[40]; const float* mlp_b2 = (const float*)d_in[41];
  const float* lno_g = (const float*)d_in[42]; const float* lno_b = (const float*)d_in[43];
  const float* pt_w = (const float*)d_in[44];
  const float* bn_g = (const float*)d_in[45]; const float* bn_b = (const float*)d_in[46];
  float* out = (float*)d_out;

  char* base = (char*)d_ws;
  size_t off = 0;
  auto alloc = [&](size_t nbytes) -> void* {
    void* p = base + off;
    off += (nbytes + 255) & ~(size_t)255;
    return p;
  };
  float* x224 = (float*)alloc((size_t)NBA * NCH * N2 * 4);
  float* A56  = (float*)alloc((size_t)NBA * NCH * 56 * 56 * 4);
  float* A112 = (float*)alloc((size_t)NBA * NCH * N1 * 4);
  float* d112 = (float*)alloc((size_t)NBA * NCH * N1 * 4);
  float* statA = (float*)alloc(4096);
  float* box0 = (float*)alloc((size_t)NBA * N1 * NCH * 4);
  float* qb   = (float*)alloc((size_t)NBA * N1 * NCH * 4);
  float* kv0  = (float*)alloc((size_t)NBA * N1 * 2 * NCH * 4);
  float* kv1  = (float*)alloc((size_t)NBA * H0 * W0 * 2 * NCH * 4);
  float* kv2  = (float*)alloc((size_t)NBA * 4 * H0 * W0 * 2 * NCH * 4);
  float* kv3  = (float*)alloc((size_t)NBA * N1 * 2 * NCH * 4);
  float* kv4  = (float*)alloc((size_t)NBA * N1 * 2 * NCH * 4);
  float* t4   = (float*)alloc((size_t)NBA * N1 * NCH * 4);
  float* ub   = (float*)alloc((size_t)NBA * N1 * NCH * 4);
  float* upb  = (float*)alloc((size_t)NBA * N1 * NCH * 4);
  float* pdo  = (float*)alloc((size_t)NBA * N1 * 4 * 4);
  float* fstat = (float*)alloc(4096);
  float* bnstat = (float*)alloc(4096);
  float* y3   = (float*)alloc((size_t)NBA * N2 * NCH * 4);
  bf16* Abf   = (bf16*)alloc((size_t)NBA * N2 * NCH * 2);
  bf16* Bbf   = (bf16*)alloc((size_t)NBA * N2 * 2 * NCH * 2);
  bf16* wbf   = (bf16*)alloc(512 * 1024);
  // phase-overlayed big buffers (regions dead by the time these are used)
  float* y  = box0;   // spans box0+qb+kv0 = exactly NBA*NCH*N2 floats
  float* y2 = kv1;    // spans kv1..kv4 (> NBA*NCH*N2 floats)
  float* z  = kv1;    // reused again after y2 is consumed

  bf16* wq_bf  = wbf;
  bf16* wkv_bf = wq_bf + 16384;
  bf16* wp_bf  = wkv_bf + 32768;
  bf16* al1_bf = wp_bf + 16384;
  bf16* al2_bf = al1_bf + 16384;
  bf16* w1_bf  = al2_bf + 16384;
  bf16* w2_bf  = w1_bf + 32768;
  bf16* pt_bf  = w2_bf + 32768;

  auto nblk = [](long n) { return (unsigned)((n + 255) / 256); };
  auto gemm = [&](const bf16* A, const bf16* W, const float* bias, float* oF,
                  bf16* oB, int M, int K, int Nn, int gelu) {
    dim3 g(Nn / 64, (M + 127) / 128);
    k_gemm<<<g, 256, 0, stream>>>(A, W, bias, oF, oB, M, K, Nn, gelu);
  };
  auto cvt = [&](const float* src, bf16* dst, long n) {
    k_cvt_bf16<<<nblk(n), 256, 0, stream>>>(src, dst, n);
  };

  // weight conversions (once)
  cvt(wq, wq_bf, 16384);       cvt(wkv, wkv_bf, 32768);
  cvt(wp, wp_bf, 16384);       cvt(al1_pw, al1_bf, 16384);
  cvt(al2_pw, al2_bf, 16384);  cvt(mlp_w1, w1_bf, 32768);
  cvt(mlp_w2, w2_bf, 32768);   cvt(pt_w, pt_bf, 16384);

  const int BC = NBA * NCH;
  // 1) pyramid fuse -> x224
  k_upbi_add<<<nblk((long)BC * 56 * 56), 256, 0, stream>>>(x1, x2, A56, BC, 28, 28);
  k_upbi_add<<<nblk((long)BC * N1), 256, 0, stream>>>(A56, x3, A112, BC, 56, 56);
  k_upbi_add<<<nblk((long)BC * N2), 256, 0, stream>>>(A112, x4, x224, BC, 112, 112);

  // 2) avg_local #1: dwconv2(x224) -> gn -> 1x1 -> box0 tokens
  k_dwconv2<<<nblk((long)BC * N1), 256, 0, stream>>>(x224, al1_dw, d112, H2, W2v);
  k_planestat<<<BC, 256, 0, stream>>>(d112, statA, N1, 0);
  k_gnapply_tok<<<nblk((long)BC * N1), 256, 0, stream>>>(d112, statA, al1_gng, al1_gnb, Abf, N1);
  gemm(Abf, al1_bf, al1_pwb, box0, nullptr, NBA * N1, NCH, NCH, 0);

  // 3) ln0 -> q, kv0
  k_rowln<<<nblk(NBA * N1), 256, 0, stream>>>(box0, ln0_g, ln0_b, nullptr, Abf, NBA * N1, NCH, 0);
  gemm(Abf, wq_bf, bq, qb, nullptr, NBA * N1, NCH, NCH, 0);
  gemm(Abf, wkv_bf, bkv, kv0, nullptr, NBA * N1, NCH, 2 * NCH, 0);

  // 4) kv1..kv3 from x1..x3 (per-pixel layernorm -> wkv)
  k_pixln<<<nblk(NBA * H0 * W0), 256, 0, stream>>>(x1, ln1_g, ln1_b, nullptr, Abf, H0 * W0, 0);
  gemm(Abf, wkv_bf, bkv, kv1, nullptr, NBA * H0 * W0, NCH, 2 * NCH, 0);
  k_pixln<<<nblk(NBA * 4 * H0 * W0), 256, 0, stream>>>(x2, ln2_g, ln2_b, nullptr, Abf, 4 * H0 * W0, 0);
  gemm(Abf, wkv_bf, bkv, kv2, nullptr, NBA * 4 * H0 * W0, NCH, 2 * NCH, 0);
  k_pixln<<<nblk(NBA * N1), 256, 0, stream>>>(x3, ln3_g, ln3_b, nullptr, Abf, N1, 0);
  gemm(Abf, wkv_bf, bkv, kv3, nullptr, NBA * N1, NCH, 2 * NCH, 0);

  // 5) avg_local #2 on x4 -> t4 -> ln4 -> kv4
  k_dwconv2<<<nblk((long)BC * N1), 256, 0, stream>>>(x4, al2_dw, d112, H2, W2v);
  k_planestat<<<BC, 256, 0, stream>>>(d112, statA, N1, 0);
  k_gnapply_tok<<<nblk((long)BC * N1), 256, 0, stream>>>(d112, statA, al2_gng, al2_gnb, Abf, N1);
  gemm(Abf, al2_bf, al2_pwb, t4, nullptr, NBA * N1, NCH, NCH, 0);
  k_rowln<<<nblk(NBA * N1), 256, 0, stream>>>(t4, ln4_g, ln4_b, nullptr, Abf, NBA * N1, NCH, 0);
  gemm(Abf, wkv_bf, bkv, kv4, nullptr, NBA * N1, NCH, 2 * NCH, 0);

  // 6) attention over 5 levels, then wp projection
  k_attn<<<nblk((long)NBA * N1 * NGR * 32), 256, 0, stream>>>(qb, kv0, kv1, kv2, kv3, kv4, ub);
  cvt(ub, Abf, (long)NBA * N1 * NCH);
  gemm(Abf, wp_bf, bp, upb, nullptr, NBA * N1, NCH, NCH, 0);

  // 7) fnorm_channel + pd + pixel-shuffle scatter-add into x224
  k_colstat<<<NBA * NCH, 256, 0, stream>>>(upb, fstat, N1, NCH, 1);
  k_pd<<<nblk((long)NBA * N1 * 32), 256, 0, stream>>>(upb, fstat, fn_std, pd_w, pd_b, pdo);
  k_scatter<<<nblk((long)BC * N1), 256, 0, stream>>>(upb, pdo, x224);

  // 8) MLP branch: lgnorm -> dw3x3 -> gn -> w1+gelu -> w2 -> lno -> residual
  k_pixln<<<nblk((long)NBA * N2), 256, 0, stream>>>(x224, gi_g, gi_b, y, nullptr, N2, 1);
  k_dwconv3<<<nblk((long)BC * N2), 256, 0, stream>>>(y, mlp_dw, y2, H2, W2v);
  k_planestat<<<BC, 256, 0, stream>>>(y2, statA, N2, 0);
  k_gnapply_tok<<<nblk((long)BC * N2), 256, 0, stream>>>(y2, statA, mlp_gng, mlp_gnb, Abf, N2);
  gemm(Abf, w1_bf, mlp_b1, nullptr, Bbf, NBA * N2, NCH, 2 * NCH, 1);
  gemm(Bbf, w2_bf, mlp_b2, y3, nullptr, NBA * N2, 2 * NCH, NCH, 0);
  k_rowln<<<nblk(NBA * N2), 256, 0, stream>>>(y3, lno_g, lno_b, y, nullptr, NBA * N2, NCH, 0);
  k_addtok<<<nblk((long)BC * N2), 256, 0, stream>>>(x224, y, N2);

  // 9) pt projection -> batchnorm -> gelu -> output (NCHW)
  k_tokbf<<<nblk((long)BC * N2), 256, 0, stream>>>(x224, Abf, N2);
  gemm(Abf, pt_bf, nullptr, z, nullptr, NBA * N2, NCH, NCH, 0);
  k_colstat<<<NCH, 256, 0, stream>>>(z, bnstat, NBA * N2, NCH, 0);
  k_bnfinal<<<nblk((long)BC * N2), 256, 0, stream>>>(z, bnstat, bn_g, bn_b, out, N2);
}
